// PatchLayer_27805618274639
// MI455X (gfx1250) — matrix operand order repaired and checked by reading, not compiled, not run on hardware
//
#include <hip/hip_runtime.h>
#include <hip/hip_bf16.h>
#include <math.h>

// ---------------------------------------------------------------------------
// Shifted-Patch Tokenization for MI455X (gfx1250, wave32, WMMA f16->f32)
//   pass0: W (3840x768 f32) -> Wt (768x3840 f16, transposed) in ws
//   pass1: images -> patches (f32, output 2) + layernormed f16 rows in ws
//   pass2: LDS-blocked WMMA GEMM: 128x128 C-tile / block, 32x64 / wave,
//          K staged in double-buffered LDS chunks via global_load_async_to_lds
// ---------------------------------------------------------------------------

typedef __attribute__((ext_vector_type(16))) _Float16 v16h;
typedef __attribute__((ext_vector_type(8)))  _Float16 v8h;
typedef __attribute__((ext_vector_type(8)))  float    v8f;
typedef __attribute__((ext_vector_type(4)))  int      v4i;

#define IMG        224
#define GRID_N     14
#define NUM_PATCH  196                   // 14*14
#define BATCH      64
#define M_TOTAL    (BATCH * NUM_PATCH)   // 12544
#define PATCH_DIM  3840                  // 16*16*15
#define PROJ_DIM   768
#define LN_EPS     1e-6f

#define KC         64                    // K chunk staged in LDS
#define NCHUNK     (PATCH_DIM / KC)      // 60
#define ROWP       72                    // padded LDS row stride (halves), 144 B

#ifndef __has_builtin
#define __has_builtin(x) 0
#endif
#if __has_builtin(__builtin_amdgcn_global_load_async_to_lds_b128)
#define HAVE_ASYNC_LDS 1
#else
#define HAVE_ASYNC_LDS 0
#endif

__device__ __forceinline__ void async_wait_all() {
#if HAVE_ASYNC_LDS
#if __has_builtin(__builtin_amdgcn_s_wait_asynccnt)
    __builtin_amdgcn_s_wait_asynccnt(0);
#else
    asm volatile("s_wait_asynccnt 0x0" ::: "memory");
#endif
#endif
}

// copy one 16-byte element per call: global -> LDS (async, ASYNCcnt-tracked)
__device__ __forceinline__ void cp16_g2l(const _Float16* g, _Float16* l) {
#if HAVE_ASYNC_LDS
    __builtin_amdgcn_global_load_async_to_lds_b128(
        (__attribute__((address_space(1))) v4i*)g,
        (__attribute__((address_space(3))) v4i*)l, 0, 0);
#else
    *(v8h*)l = *(const v8h*)g;
#endif
}

__device__ __forceinline__ v16h ld_frag(const _Float16* p, int hi_off_halves) {
    v8h lo = *(const v8h*)p;
    v8h hi = *(const v8h*)(p + hi_off_halves);
    return __builtin_shufflevector(lo, hi,
           0,1,2,3,4,5,6,7,8,9,10,11,12,13,14,15);
}

// -------------------------- pass 0: W transpose + f16 ----------------------
__global__ __launch_bounds__(256)
void spt_convert_w(const float* __restrict__ W, _Float16* __restrict__ Wt) {
    int idx = blockIdx.x * 256 + threadIdx.x;          // over 768*3840
    if (idx >= PROJ_DIM * PATCH_DIM) return;
    int n = idx / PATCH_DIM;
    int k = idx - n * PATCH_DIM;
    Wt[idx] = (_Float16)W[(size_t)k * PROJ_DIM + n];   // Wt[n][k] = W[k][n]
}

// ---------------- pass 1: patch extract + layernorm (one block per row) ----
__global__ __launch_bounds__(256)
void spt_patch_ln(const float* __restrict__ img,
                  const float* __restrict__ gamma,
                  const float* __restrict__ beta,
                  float* __restrict__ patches_out,
                  _Float16* __restrict__ normed) {
    const int pid = blockIdx.x;              // 0..12543 == b*196 + gy*14 + gx
    const int b   = pid / NUM_PATCH;
    const int pg  = pid - b * NUM_PATCH;
    const int gy  = pg / GRID_N;
    const int gx  = pg - gy * GRID_N;
    const int t   = threadIdx.x;

    // concat groups: [identity, (+8,+8), (-8,+8), (+8,-8), (-8,-8)], zero-fill OOB
    const int dy[5] = {0,  8, -8,  8, -8};
    const int dx[5] = {0,  8,  8, -8, -8};

    float val[15];
    float sum = 0.f, sq = 0.f;
#pragma unroll
    for (int i = 0; i < 15; ++i) {
        int d   = t + (i << 8);              // d-strided: coalesced stores later
        int pix = d / 15;
        int ch  = d - pix * 15;
        int g   = ch / 3;
        int c   = ch - g * 3;
        int py  = pix >> 4, px = pix & 15;
        int y   = gy * 16 + py + dy[g];
        int x   = gx * 16 + px + dx[g];
        float v = 0.f;
        if ((unsigned)y < (unsigned)IMG && (unsigned)x < (unsigned)IMG)
            v = img[(((size_t)b * IMG + y) * IMG + x) * 3 + c];
        val[i] = v;
        sum += v;
        sq  += v * v;
    }

    __shared__ float ssum[256];
    __shared__ float ssq[256];
    ssum[t] = sum; ssq[t] = sq;
    __syncthreads();
#pragma unroll
    for (int o = 128; o > 0; o >>= 1) {
        if (t < o) { ssum[t] += ssum[t + o]; ssq[t] += ssq[t + o]; }
        __syncthreads();
    }
    __shared__ float s_mean, s_rstd;
    if (t == 0) {
        float m = ssum[0] * (1.f / PATCH_DIM);
        float v = ssq[0] * (1.f / PATCH_DIM) - m * m;
        s_mean = m;
        s_rstd = rsqrtf(v + LN_EPS);
    }
    __syncthreads();
    const float mean = s_mean, rstd = s_rstd;

    const size_t row = (size_t)pid * PATCH_DIM;
#pragma unroll
    for (int i = 0; i < 15; ++i) {
        int d = t + (i << 8);
        __builtin_nontemporal_store(val[i], &patches_out[row + d]); // streamed output
        float nv = (val[i] - mean) * rstd * gamma[d] + beta[d];
        normed[row + d] = (_Float16)nv;                             // keep in L2
    }
}

// ------- pass 2: LDS-blocked WMMA GEMM (128x128 / block, 32x64 / wave) -----
__global__ __launch_bounds__(256)
void spt_gemm(const _Float16* __restrict__ A,    // [12544][3840] f16 row-major
              const _Float16* __restrict__ Bt,   // [768][3840]   Wt
              const float*    __restrict__ bias, // [768]
              float*          __restrict__ out)  // [12544][768]
{
    __shared__ _Float16 sA[2][128 * ROWP];       // 36 KB
    __shared__ _Float16 sB[2][128 * ROWP];       // 36 KB

    const int tid  = threadIdx.x;
    const int lane = tid & 31;
    const int wid  = tid >> 5;        // 0..7
    const int wm   = wid & 3;         // M sub-group: 32 rows each
    const int wn   = wid >> 2;        // N sub-group: 64 cols each
    const int r    = lane & 15;
    const int hs   = lane >> 4;       // half-wave select

    const int m_block = blockIdx.x * 128;
    const int n_block = blockIdx.y * 128;

    // --- cooperative copy assignment: 4 b128 for A + 4 b128 for B per thread
    // idx = tid + i*256 -> row = (tid>>3) + 32*i, col (halves) = (tid&7)*8
    const int crow = tid >> 3;
    const int ccol = (tid & 7) * 8;
    const _Float16* gA = A  + (size_t)(m_block + crow) * PATCH_DIM + ccol;
    const _Float16* gB = Bt + (size_t)(n_block + crow) * PATCH_DIM + ccol;
    const int loff = crow * ROWP + ccol;

    v8f acc[2][4] = {};

    // prologue: stage chunk 0 into buffer 0
#pragma unroll
    for (int i = 0; i < 4; ++i) {
        cp16_g2l(gA + (size_t)i * 32 * PATCH_DIM, &sA[0][loff + i * 32 * ROWP]);
        cp16_g2l(gB + (size_t)i * 32 * PATCH_DIM, &sB[0][loff + i * 32 * ROWP]);
    }
    async_wait_all();
    __syncthreads();

    for (int c = 0; c < NCHUNK; ++c) {
        // issue next chunk's copies into the other buffer (overlaps compute)
        if (c + 1 < NCHUNK) {
            const int nb = (c + 1) & 1;
            const size_t k0 = (size_t)(c + 1) * KC;
#pragma unroll
            for (int i = 0; i < 4; ++i) {
                cp16_g2l(gA + k0 + (size_t)i * 32 * PATCH_DIM,
                         &sA[nb][loff + i * 32 * ROWP]);
                cp16_g2l(gB + k0 + (size_t)i * 32 * PATCH_DIM,
                         &sB[nb][loff + i * 32 * ROWP]);
            }
        }

        // compute on current buffer: 2 k-steps x 8 WMMA
        const _Float16* As = sA[c & 1];
        const _Float16* Bs = sB[c & 1];
#pragma unroll
        for (int kk = 0; kk < KC; kk += 32) {
            // A frags (16-bit 16x32 layout): row = m_local + r;
            //   halves[0..8)=K kk+8*hs, halves[8..16)=K kk+16+8*hs
            const _Float16* a0p = As + (wm * 32 + r) * ROWP + kk + hs * 8;
            v16h a0 = ld_frag(a0p, 16);
            v16h a1 = ld_frag(a0p + 16 * ROWP, 16);
#pragma unroll
            for (int t = 0; t < 4; ++t) {
                // B frag (32x16 layout): col N = n_local + r; K kk+8*hs.., kk+16+8*hs..
                const _Float16* bp =
                    Bs + (wn * 64 + t * 16 + r) * ROWP + kk + hs * 8;
                v16h bf = ld_frag(bp, 16);
                acc[0][t] = __builtin_amdgcn_wmma_f32_16x16x32_f16(
                    false, a0, false, bf, (short)0, acc[0][t], false, false);
                acc[1][t] = __builtin_amdgcn_wmma_f32_16x16x32_f16(
                    false, a1, false, bf, (short)0, acc[1][t], false, false);
            }
        }

        async_wait_all();       // next chunk landed in LDS (this wave's copies)
        __syncthreads();        // all waves done: safe to read next / refill cur
    }

    // epilogue: C/D layout — VGPR j holds (M = base + j + 8*hs, N = base + r)
#pragma unroll
    for (int g = 0; g < 2; ++g) {
#pragma unroll
        for (int t = 0; t < 4; ++t) {
            int n = n_block + wn * 64 + t * 16 + r;
            float bv = bias[n];
            int m0 = m_block + wm * 32 + g * 16 + hs * 8;
#pragma unroll
            for (int j = 0; j < 8; ++j)
                __builtin_nontemporal_store(acc[g][t][j] + bv,
                    &out[(size_t)(m0 + j) * PROJ_DIM + n]);
        }
    }
}

// ---------------------------------------------------------------------------
extern "C" void kernel_launch(void* const* d_in, const int* in_sizes, int n_in,
                              void* d_out, int out_size, void* d_ws, size_t ws_size,
                              hipStream_t stream) {
    const float* images = (const float*)d_in[0]; // (64,224,224,3)
    const float* gamma  = (const float*)d_in[1]; // (3840,)
    const float* beta   = (const float*)d_in[2]; // (3840,)
    const float* W      = (const float*)d_in[3]; // (3840,768)
    const float* bias   = (const float*)d_in[4]; // (768,)

    float* tokens  = (float*)d_out;                                   // 12544*768
    float* patches = (float*)d_out + (size_t)M_TOTAL * PROJ_DIM;      // 12544*3840

    // workspace: Wt f16 (5.9 MB) then normed f16 (96.3 MB)
    _Float16* Wt     = (_Float16*)d_ws;
    _Float16* normed = (_Float16*)((char*)d_ws +
                        (size_t)PROJ_DIM * PATCH_DIM * sizeof(_Float16));

    // pass 0: transpose+convert W
    {
        int total = PROJ_DIM * PATCH_DIM;
        spt_convert_w<<<(total + 255) / 256, 256, 0, stream>>>(W, Wt);
    }
    // pass 1: patches + layernorm (one block per token row)
    spt_patch_ln<<<M_TOTAL, 256, 0, stream>>>(images, gamma, beta, patches, normed);

    // pass 2: LDS-blocked WMMA GEMM; 98 x 6 blocks of 128x128
    {
        dim3 grid(M_TOTAL / 128, PROJ_DIM / 128);   // (98, 6)
        spt_gemm<<<grid, 256, 0, stream>>>(normed, Wt, bias, tokens);
    }
}
